// PLCCLoss_86354612453531
// MI455X (gfx1250) — matrix operand order repaired and checked by reading, not compiled, not run on hardware
//
#include <hip/hip_runtime.h>
#include <stdint.h>

// ---------------------------------------------------------------------------
// PLCC loss on MI455X (gfx1250): bf16x2-split WMMA GEMM for cosine similarity,
// TDM (tensor_load_to_lds) double-buffered B staging, fused pairwise epilogue.
// ---------------------------------------------------------------------------

typedef __attribute__((ext_vector_type(16))) __bf16 v16bf;
typedef __attribute__((ext_vector_type(8)))  __bf16 v8bf;
typedef __attribute__((ext_vector_type(8)))  float  v8f;
typedef __attribute__((ext_vector_type(4)))  unsigned int v4u;
typedef __attribute__((ext_vector_type(8)))  int v8i;
typedef __attribute__((ext_vector_type(4)))  int v4i;

#define NPTS 8192
#define DIM  64
#define NTJ  (NPTS / 16)      // 512 j-tiles
#define INV_TEMP 10.0f        // 1/0.1
#define GAMMA 0.5f

// ---------------------------------------------------------------------------
// Kernel 1: normalize rows -> bf16 hi/lo split; coord sq-norms; zero accums.
// ---------------------------------------------------------------------------
__global__ __launch_bounds__(256) void plcc_prep_kernel(
    const float* __restrict__ feat, const float* __restrict__ coords,
    __bf16* __restrict__ fnhi, __bf16* __restrict__ fnlo,
    float* __restrict__ sqn, float* __restrict__ rowPos,
    float* __restrict__ rowNeg, float* __restrict__ contAcc)
{
    int i = blockIdx.x * 256 + threadIdx.x;
    if (i == 0) *contAcc = 0.0f;
    if (i >= NPTS) return;

    const float* f = feat + (size_t)i * DIM;
    float s = 0.0f;
#pragma unroll 8
    for (int k = 0; k < DIM; ++k) s += f[k] * f[k];
    float nrm = fmaxf(sqrtf(s), 1e-8f);
    float inv = 1.0f / nrm;

#pragma unroll 8
    for (int k = 0; k < DIM; ++k) {
        float x = f[k] * inv;
        __bf16 h = (__bf16)x;
        float r  = x - (float)h;
        fnhi[(size_t)i * DIM + k] = h;
        fnlo[(size_t)i * DIM + k] = (__bf16)r;
    }

    float cx = coords[3 * i + 0];
    float cy = coords[3 * i + 1];
    float cz = coords[3 * i + 2];
    sqn[i]    = cx * cx + cy * cy + cz * cz;
    rowPos[i] = 0.0f;
    rowNeg[i] = 0.0f;
}

// ---------------------------------------------------------------------------
// TDM: DMA one 16-row x 128-byte tile (16x64 bf16) from global into LDS.
// 2D descriptor, data_size = 8B units: tile_dim0 = 16 units/row, tile_dim1 = 16
// rows, tensor_dim0_stride = 16 units (contiguous rows). Groups 2/3 unused.
// ---------------------------------------------------------------------------
__device__ __forceinline__ void tdm_load_tile16x64(unsigned lds_byte_off,
                                                   const __bf16* gsrc)
{
    unsigned long long ga = (unsigned long long)(uintptr_t)gsrc;
    v4u g0;
    g0[0] = 1u;                                   // count=1, user-mode, no gather
    g0[1] = lds_byte_off;                         // lds_addr
    g0[2] = (unsigned)(ga & 0xFFFFFFFFull);       // global_addr[31:0]
    g0[3] = (unsigned)((ga >> 32) & 0x01FFFFFFull) | (2u << 30); // addr[56:32], type=2
    v8i g1;
    g1[0] = (int)(3u << 16);       // wg_mask=0, data_size=3 (8B), no pad/iterate
    g1[1] = (int)(16u << 16);      // tensor_dim0 = 16 units (128B row)
    g1[2] = (int)(0x2000u << 16);  // tensor_dim0 hi=0 | tensor_dim1 = 8192 rows
    g1[3] = (int)(16u << 16);      // tensor_dim1 hi=0 | tile_dim0 = 16 units
    g1[4] = 16;                    // tile_dim1 = 16 rows, tile_dim2 = 0
    g1[5] = 16;                    // tensor_dim0_stride = 16 units
    g1[6] = 0;
    g1[7] = 0;
    v4i gz4 = {0, 0, 0, 0};
    v8i gz8 = {0, 0, 0, 0, 0, 0, 0, 0};
    __builtin_amdgcn_tensor_load_to_lds(g0, g1, gz4, gz4, gz8, 0);
}

// ---------------------------------------------------------------------------
// Kernel 2: pairwise tile kernel. 8 waves/WG, each wave owns a 16-row i-strip,
// loops over j-tiles. sim tile via 6x v_wmma_f32_16x16x32_bf16 (bf16x2 split),
// fused exp/mask/distance epilogue (branchless, HW sqrt/exp TRANS ops),
// in-register row accumulation across the whole j loop.
// ---------------------------------------------------------------------------
__global__ __launch_bounds__(256) void plcc_pair_kernel(
    const int* __restrict__ labels, const float* __restrict__ coords,
    const __bf16* __restrict__ fnhi, const __bf16* __restrict__ fnlo,
    const float* __restrict__ sqn,
    float* __restrict__ rowPos, float* __restrict__ rowNeg,
    float* __restrict__ contAcc)
{
    __shared__ __align__(32) __bf16 Bhi[2][16 * DIM];
    __shared__ __align__(32) __bf16 Blo[2][16 * DIM];

    const int lane = threadIdx.x & 31;
    const int wave = threadIdx.x >> 5;
    const int grp  = lane >> 4;     // K-half / M-half selector
    const int l15  = lane & 15;
    const int i0   = blockIdx.x * 128 + wave * 16;

    // ---- A operands in registers (16x32 tile, contiguous k order, same as B):
    //      lane l: row = i0 + (l&15); V0-7 = K[16*grp .. +15]
    const __bf16* ahrow = fnhi + (size_t)(i0 + l15) * DIM;
    const __bf16* alrow = fnlo + (size_t)(i0 + l15) * DIM;
    v16bf Ah[2], Al[2];
#pragma unroll
    for (int s = 0; s < 2; ++s) {
        int base = s * 32 + grp * 16;
        v8bf a0 = *(const v8bf*)(ahrow + base);
        v8bf a1 = *(const v8bf*)(ahrow + base + 8);
        Ah[s] = __builtin_shufflevector(a0, a1, 0,1,2,3,4,5,6,7,8,9,10,11,12,13,14,15);
        v8bf b0 = *(const v8bf*)(alrow + base);
        v8bf b1 = *(const v8bf*)(alrow + base + 8);
        Al[s] = __builtin_shufflevector(b0, b1, 0,1,2,3,4,5,6,7,8,9,10,11,12,13,14,15);
    }

    // ---- per-i-row metadata (loop invariant): C slot v -> row i0 + v + 8*grp
    float sqni[8], cix[8], ciy[8], ciz[8];
    bool  mi[8];
#pragma unroll
    for (int v = 0; v < 8; ++v) {
        int i = i0 + v + 8 * grp;
        mi[v]   = (labels[i] == 2);
        sqni[v] = sqn[i];
        cix[v] = coords[3 * i + 0];
        ciy[v] = coords[3 * i + 1];
        ciz[v] = coords[3 * i + 2];
    }

    float accP[8] = {0, 0, 0, 0, 0, 0, 0, 0};
    float accN[8] = {0, 0, 0, 0, 0, 0, 0, 0};
    float cont = 0.0f;

    const int step = gridDim.y;
    int jt = blockIdx.y;
    int buf = 0;

    // prologue DMA (wave 0 drives the TDM; barrier publishes to all waves)
    if (wave == 0) {
        tdm_load_tile16x64((unsigned)(size_t)&Bhi[0][0], fnhi + (size_t)jt * 16 * DIM);
        tdm_load_tile16x64((unsigned)(size_t)&Blo[0][0], fnlo + (size_t)jt * 16 * DIM);
    }

    for (; jt < NTJ; jt += step) {
        if (wave == 0) __builtin_amdgcn_s_wait_tensorcnt(0);
        __syncthreads();                       // current buffer ready for all waves

        int jn = jt + step;
        if (wave == 0 && jn < NTJ) {           // prefetch next tile into other buffer
            tdm_load_tile16x64((unsigned)(size_t)&Bhi[buf ^ 1][0], fnhi + (size_t)jn * 16 * DIM);
            tdm_load_tile16x64((unsigned)(size_t)&Blo[buf ^ 1][0], fnlo + (size_t)jn * 16 * DIM);
        }

        // ---- B operands from LDS (32x16 tile, contiguous k order, same as A):
        //      lane l: col = l&15 -> row j0+(l&15); K = 32*s + 16*grp .. +15
        v16bf Bh[2], Bl[2];
#pragma unroll
        for (int s = 0; s < 2; ++s) {
            int off = l15 * DIM + s * 32 + grp * 16;
            Bh[s] = *(const v16bf*)&Bhi[buf][off];
            Bl[s] = *(const v16bf*)&Blo[buf][off];
        }

        // ---- sim = hi*hi + hi*lo + lo*hi (small terms first), K=64 in 2 steps
        v8f c = {0, 0, 0, 0, 0, 0, 0, 0};
#pragma unroll
        for (int s = 0; s < 2; ++s) {
            c = __builtin_amdgcn_wmma_f32_16x16x32_bf16(false, Al[s], false, Bh[s],
                                                        (short)0, c, false, false);
            c = __builtin_amdgcn_wmma_f32_16x16x32_bf16(false, Ah[s], false, Bl[s],
                                                        (short)0, c, false, false);
            c = __builtin_amdgcn_wmma_f32_16x16x32_bf16(false, Ah[s], false, Bh[s],
                                                        (short)0, c, false, false);
        }

        // ---- fused epilogue: exp / masks / distances / accumulation (branchless)
        int j = jt * 16 + l15;
        bool  mj   = (labels[j] == 2);
        float sqnj = sqn[j];
        float cjx = coords[3 * j + 0];
        float cjy = coords[3 * j + 1];
        float cjz = coords[3 * j + 2];

#pragma unroll
        for (int v = 0; v < 8; ++v) {
            float sim = c[v];
            float e   = __expf(sim * INV_TEMP);          // v_exp_f32 path
            float sq  = sqni[v] + sqnj
                      - 2.0f * (cix[v] * cjx + ciy[v] * cjy + ciz[v] * cjz);
            sq = fmaxf(sq, 0.0f);
            float dist = __builtin_amdgcn_sqrtf(sq);     // single v_sqrt_f32
            bool pairm  = mi[v] && mj;
            bool within = (sq < 1.0f) && (sq > 1e-12f);
            bool pos = pairm && within;
            accP[v] += pos ? e : 0.0f;
            accN[v] += (pairm && !within) ? e : 0.0f;
            cont    += pos ? fabsf((1.0f - sim) - dist) : 0.0f;
        }

        __syncthreads();                       // all waves done reading buf
        buf ^= 1;
    }

    // ---- row sums: reduce over the 16 lanes sharing a C row (masks stay in-half)
#pragma unroll
    for (int v = 0; v < 8; ++v) {
        float p = accP[v], n = accN[v];
        for (int m = 8; m >= 1; m >>= 1) {
            p += __shfl_xor(p, m, 32);
            n += __shfl_xor(n, m, 32);
        }
        if (l15 == 0) {
            int i = i0 + v + 8 * grp;
            atomicAdd(&rowPos[i], p);
            atomicAdd(&rowNeg[i], n);
        }
    }
    for (int m = 16; m >= 1; m >>= 1) cont += __shfl_xor(cont, m, 32);
    if (lane == 0) atomicAdd(contAcc, cont);
}

// ---------------------------------------------------------------------------
// Kernel 3: final scalar reduction.
// ---------------------------------------------------------------------------
__global__ __launch_bounds__(1024) void plcc_final_kernel(
    const int* __restrict__ labels, const float* __restrict__ rowPos,
    const float* __restrict__ rowNeg, const float* __restrict__ contAcc,
    float* __restrict__ out)
{
    __shared__ float sInfo[1024];
    __shared__ float sL[1024];
    int t = threadIdx.x;
    float info = 0.0f, L = 0.0f;
    for (int i = t; i < NPTS; i += 1024) {
        if (labels[i] == 2) {
            float p = rowPos[i], n = rowNeg[i];
            float ratio = p / (n + p + 1e-6f);
            info += -__logf(ratio);
            L += 1.0f;
        }
    }
    sInfo[t] = info;
    sL[t]    = L;
    __syncthreads();
    for (int ofs = 512; ofs > 0; ofs >>= 1) {
        if (t < ofs) {
            sInfo[t] += sInfo[t + ofs];
            sL[t]    += sL[t + ofs];
        }
        __syncthreads();
    }
    if (t == 0) {
        float Lf = sL[0];
        out[0] = sInfo[0] / Lf + GAMMA * (*contAcc) / (Lf * Lf);
    }
}

// ---------------------------------------------------------------------------
extern "C" void kernel_launch(void* const* d_in, const int* in_sizes, int n_in,
                              void* d_out, int out_size, void* d_ws, size_t ws_size,
                              hipStream_t stream)
{
    (void)in_sizes; (void)n_in; (void)out_size; (void)ws_size;
    const float* features = (const float*)d_in[0];
    const int*   labels   = (const int*)d_in[1];
    const float* coords   = (const float*)d_in[2];
    float* out = (float*)d_out;

    char* ws = (char*)d_ws;
    __bf16* fnhi   = (__bf16*)(ws);                          // 1 MB
    __bf16* fnlo   = (__bf16*)(ws + (1u << 20));             // 1 MB
    float*  sqn    = (float*)(ws + (2u << 20));              // 32 KB
    float*  rowPos = (float*)(ws + (2u << 20) + 32768);      // 32 KB
    float*  rowNeg = (float*)(ws + (2u << 20) + 65536);      // 32 KB
    float*  cont   = (float*)(ws + (2u << 20) + 98304);      // 4 B

    plcc_prep_kernel<<<dim3(NPTS / 256), dim3(256), 0, stream>>>(
        features, coords, fnhi, fnlo, sqn, rowPos, rowNeg, cont);
    plcc_pair_kernel<<<dim3(NPTS / 128, 8), dim3(256), 0, stream>>>(
        labels, coords, fnhi, fnlo, sqn, rowPos, rowNeg, cont);
    plcc_final_kernel<<<dim3(1), dim3(1024), 0, stream>>>(
        labels, rowPos, rowNeg, cont, out);
}
